// RW_Module_72018011619946
// MI455X (gfx1250) — compile-verified, matrix operand order checked by reading, not run on hardware
//
#include <hip/hip_runtime.h>
#include <hip/hip_bf16.h>

#define BATCH 4
#define CH    512
#define NTOK  4096   // 64*64
#define DQK   64

typedef __attribute__((ext_vector_type(16))) _Float16 v16h;
typedef __attribute__((ext_vector_type(8)))  _Float16 v8h;
typedef __attribute__((ext_vector_type(8)))  float    v8f;
typedef __attribute__((ext_vector_type(4)))  float    v4f;

// Load a 16x32 f16 fragment (A-matrix layout; identical lane pattern serves as
// the B-matrix fragment when B is the transpose of a row-major matrix).
// CDNA5 16-bit A 16x32 layout: lanes 0-15 hold row (lane), K halves [0..7],[16..23];
// lanes 16-31 hold row (lane-16), K halves [8..15],[24..31].
__device__ __forceinline__ v16h load_frag_16x32(const _Float16* __restrict__ base,
                                                int ld, int lane) {
  const int r    = lane & 15;
  const int koff = (lane & 16) ? 8 : 0;
  const _Float16* row = base + (size_t)r * ld + koff;
  v8h lo = *(const v8h*)(row);        // K = koff .. koff+7
  v8h hi = *(const v8h*)(row + 16);   // K = koff+16 .. koff+23
  v16h f;
#pragma unroll
  for (int t = 0; t < 8; ++t) { f[t] = lo[t]; f[t + 8] = hi[t]; }
  return f;
}

// A fragment of X^T (16 tokens x 32 channels): element (n, c) = x[b, c, n].
// Per-lane dword gathers; lanes 0-15 cover 16 contiguous tokens -> 2 lines/load.
__device__ __forceinline__ v16h load_xt_frag(const float* __restrict__ xb,
                                             int k0, int nl, int cofs) {
  v16h af;
#pragma unroll
  for (int t = 0; t < 8; ++t) {
    af[t]     = (_Float16)xb[(size_t)(k0 + cofs + t) * NTOK + nl];
    af[t + 8] = (_Float16)xb[(size_t)(k0 + 16 + cofs + t) * NTOK + nl];
  }
  return af;
}

// B fragment from a row-major f32 weight matrix (W row per lane, f32 -> f16).
__device__ __forceinline__ v16h load_w_frag(const float* __restrict__ wrow, int k0, int cofs) {
  const v4f* wp = (const v4f*)(wrow + k0 + cofs);
  const v4f w0 = wp[0], w1 = wp[1];              // c = k0+cofs .. +7
  const v4f w2 = wp[4], w3 = wp[5];              // c = k0+cofs+16 .. +23
  v16h bf;
#pragma unroll
  for (int e = 0; e < 4; ++e) {
    bf[e]      = (_Float16)w0[e];
    bf[e + 4]  = (_Float16)w1[e];
    bf[e + 8]  = (_Float16)w2[e];
    bf[e + 12] = (_Float16)w3[e];
  }
  return bf;
}

// ---------------------------------------------------------------------------
// Q/K projection: Out^T[n, 0..127] = X^T x [Wq;Wk]^T + [bq;bk].
// One wave = 16 tokens x 128 outputs (8 tiles; t<4 -> Q, t>=4 -> K, all
// selection compile-time). Epilogue: LDS 16x16 tile transpose so global
// stores are contiguous 16B chunks instead of 2B scatters.
// ---------------------------------------------------------------------------
__global__ __launch_bounds__(128) void proj_qk_wmma(
    const float* __restrict__ x,
    const float* __restrict__ Wq, const float* __restrict__ bq,
    const float* __restrict__ Wk, const float* __restrict__ bk,
    _Float16* __restrict__ Qh, _Float16* __restrict__ Kh) {
  __shared__ _Float16 lds[4][8][16][16];
  const int lane = threadIdx.x & 31;
  const int wave = threadIdx.x >> 5;                 // 0..3
  const int n0   = (blockIdx.x * 4 + wave) * 16;     // token tile
  const int b    = blockIdx.y;

  const float* __restrict__ xb = x + (size_t)b * CH * NTOK;
  const int nl   = n0 + (lane & 15);
  const int cofs = (lane & 16) ? 8 : 0;
  const int ocol = lane & 15;

  // Hoisted W row pointers (o = t*16 + ocol)
  const float* wrow[8];
#pragma unroll
  for (int t = 0; t < 4; ++t) wrow[t] = Wq + (size_t)(t * 16 + ocol) * CH;
#pragma unroll
  for (int t = 4; t < 8; ++t) wrow[t] = Wk + (size_t)((t - 4) * 16 + ocol) * CH;

  v8f acc[8];
  const v8f vzero = {};
#pragma unroll
  for (int t = 0; t < 8; ++t) acc[t] = vzero;

  for (int k0 = 0; k0 < CH; k0 += 32) {
    const v16h af = load_xt_frag(xb, k0, nl, cofs);
#pragma unroll
    for (int t = 0; t < 8; ++t) {
      const v16h bf = load_w_frag(wrow[t], k0, cofs);
      acc[t] = __builtin_amdgcn_wmma_f32_16x16x32_f16(false, af, false, bf,
                                                      (short)0, acc[t], false, false);
    }
  }

  // Phase 1: bias + tiles -> LDS as [n_local][o_local] (contiguous lane writes)
  const int rbase = (lane & 16) ? 8 : 0;             // n_local base from lane half
#pragma unroll
  for (int t = 0; t < 8; ++t) {
    const float bias = (t < 4) ? bq[t * 16 + ocol] : bk[(t - 4) * 16 + ocol];
#pragma unroll
    for (int r = 0; r < 8; ++r)
      lds[wave][t][rbase + r][ocol] = (_Float16)(acc[t][r] + bias);
  }
  __syncthreads();

  // Phase 2: one row (32B) per lane, contiguous global_store_b128 x2
#pragma unroll
  for (int iter = 0; iter < 4; ++iter) {
    const int tile = iter * 2 + (lane >> 4);
    const int nr   = lane & 15;
    const v8h d0 = *(const v8h*)&lds[wave][tile][nr][0];
    const v8h d1 = *(const v8h*)&lds[wave][tile][nr][8];
    _Float16* __restrict__ dst;
    if (iter < 2) dst = Qh + ((size_t)b * NTOK + n0 + nr) * DQK + tile * 16;
    else          dst = Kh + ((size_t)b * NTOK + n0 + nr) * DQK + (tile - 4) * 16;
    *(v8h*)dst = d0;
    *(v8h*)(dst + 8) = d1;
  }
}

// ---------------------------------------------------------------------------
// V projection: Vt[c, n] = (X^T x Wv^T + bv)^T, channel-major output.
// One wave = 16 tokens x 128 V channels; same LDS-transpose epilogue, staged
// as [c_local][n_local] so Vt rows are written as contiguous 16B chunks.
// ---------------------------------------------------------------------------
__global__ __launch_bounds__(128) void proj_v_wmma(
    const float* __restrict__ x,
    const float* __restrict__ Wv, const float* __restrict__ bv,
    _Float16* __restrict__ Vt) {
  __shared__ _Float16 lds[4][8][16][16];
  const int lane = threadIdx.x & 31;
  const int wave = threadIdx.x >> 5;                 // 0..3
  const int n0   = (blockIdx.x * 4 + wave) * 16;     // token tile
  const int o0   = blockIdx.y * 128;                 // V-channel slab (4 slabs)
  const int b    = blockIdx.z;

  const float* __restrict__ xb = x + (size_t)b * CH * NTOK;
  const int nl   = n0 + (lane & 15);
  const int cofs = (lane & 16) ? 8 : 0;
  const int ocol = lane & 15;

  const float* wrow[8];
#pragma unroll
  for (int t = 0; t < 8; ++t) wrow[t] = Wv + (size_t)(o0 + t * 16 + ocol) * CH;

  v8f acc[8];
  const v8f vzero = {};
#pragma unroll
  for (int t = 0; t < 8; ++t) acc[t] = vzero;

  for (int k0 = 0; k0 < CH; k0 += 32) {
    const v16h af = load_xt_frag(xb, k0, nl, cofs);
#pragma unroll
    for (int t = 0; t < 8; ++t) {
      const v16h bf = load_w_frag(wrow[t], k0, cofs);
      acc[t] = __builtin_amdgcn_wmma_f32_16x16x32_f16(false, af, false, bf,
                                                      (short)0, acc[t], false, false);
    }
  }

  // Phase 1: tiles -> LDS as [c_local][n_local]... note D tile has lane = o(c),
  // VGPR = n, so to land [c][n] rows we stage transposed: index [o_local][n..]
  // Wait: we want Vt rows (fixed c, contiguous n). D tile: lane -> c, VGPR -> n.
  // Store LDS[c_local][n_local]: c_local = ocol (lane), n_local = rbase + r.
  const int rbase = (lane & 16) ? 8 : 0;
#pragma unroll
  for (int t = 0; t < 8; ++t) {
    const float bias = bv[o0 + t * 16 + ocol];
#pragma unroll
    for (int r = 0; r < 8; ++r)
      lds[wave][t][ocol][rbase + r] = (_Float16)(acc[t][r] + bias);
  }
  __syncthreads();

  // Phase 2: one c-row (32B, 16 tokens) per lane -> contiguous stores
#pragma unroll
  for (int iter = 0; iter < 4; ++iter) {
    const int tile = iter * 2 + (lane >> 4);
    const int cr   = lane & 15;
    const v8h d0 = *(const v8h*)&lds[wave][tile][cr][0];
    const v8h d1 = *(const v8h*)&lds[wave][tile][cr][8];
    const int c = o0 + tile * 16 + cr;
    _Float16* __restrict__ dst = Vt + ((size_t)b * CH + c) * NTOK + n0;
    *(v8h*)dst = d0;
    *(v8h*)(dst + 8) = d1;
  }
}

// ---------------------------------------------------------------------------
// Flash attention (transposed formulation), one wave = 16 queries x 128 chans,
// 64 keys per outer step (24 WMMAs per step):
//   S^T  = K_tile (A) x Q^T (B)          -> lane = query i, VGPR = key j
//   P^T  = exp(S^T - m)                  -> repacks into B fragments in-lane
//   O^T += V^T_tile (A) x P^T (B)        -> lane = query i, VGPR = channel c
// Online softmax state is per-lane; lane L and L^16 share query row i, merged
// with one shfl_xor(16) per step (max) and once at the end (sum).
// ---------------------------------------------------------------------------
__global__ __launch_bounds__(128) void attn_flash_kernel(
    const _Float16* __restrict__ Qh, const _Float16* __restrict__ Kh,
    const _Float16* __restrict__ Vt, const float* __restrict__ x,
    const float* __restrict__ gamma, float* __restrict__ out) {
  const int lane = threadIdx.x & 31;
  const int wave = threadIdx.x >> 5;     // 0..3
  const int i0   = blockIdx.x * 16;      // query tile
  const int b    = blockIdx.y;
  const int c0   = wave * 128;           // this wave's channel slab

  const _Float16* __restrict__ Qb = Qh + (size_t)b * NTOK * DQK;
  const _Float16* __restrict__ Kb = Kh + (size_t)b * NTOK * DQK;
  const _Float16* __restrict__ Vb = Vt + (size_t)b * CH * NTOK;

  // Q^T B-fragments for d = 0..31 and 32..63 (reused for all 4096 keys)
  const v16h qf0 = load_frag_16x32(Qb + (size_t)i0 * DQK + 0,  DQK, lane);
  const v16h qf1 = load_frag_16x32(Qb + (size_t)i0 * DQK + 32, DQK, lane);

  v8f acc[8];
  const v8f vzero = {};
#pragma unroll
  for (int t = 0; t < 8; ++t) acc[t] = vzero;
  float m_run = -1e30f;
  float l_run = 0.0f;

  for (int j0 = 0; j0 < NTOK; j0 += 64) {
    // ---- S^T: four 16x16 f32 tiles covering keys j0 .. j0+63 ----
    v8f st[4];
#pragma unroll
    for (int s = 0; s < 4; ++s) {
      const v16h kf0 = load_frag_16x32(Kb + (size_t)(j0 + 16 * s) * DQK + 0,  DQK, lane);
      const v16h kf1 = load_frag_16x32(Kb + (size_t)(j0 + 16 * s) * DQK + 32, DQK, lane);
      v8f sv = vzero;
      sv = __builtin_amdgcn_wmma_f32_16x16x32_f16(false, kf0, false, qf0, (short)0, sv, false, false);
      sv = __builtin_amdgcn_wmma_f32_16x16x32_f16(false, kf1, false, qf1, (short)0, sv, false, false);
      st[s] = sv;
    }

    // ---- online softmax (per-lane; merge row halves via lane^16) ----
    float mb = st[0][0];
#pragma unroll
    for (int s = 0; s < 4; ++s)
#pragma unroll
      for (int t = 0; t < 8; ++t) mb = fmaxf(mb, st[s][t]);
    mb = fmaxf(mb, __shfl_xor(mb, 16, 32));
    const float m_new = fmaxf(m_run, mb);
    const float scale = __expf(m_run - m_new);
    m_run = m_new;

    // P^T = exp(S^T - m_new); pack straight into two 32x16 B fragments:
    // st[0]/st[1] -> pf0 K 0..31, st[2]/st[3] -> pf1 K 32..63; no lane moves.
    v16h pf0, pf1;
    float psum = 0.0f;
#pragma unroll
    for (int t = 0; t < 8; ++t) {
      const float p0 = __expf(st[0][t] - m_new);
      const float p1 = __expf(st[1][t] - m_new);
      const float p2 = __expf(st[2][t] - m_new);
      const float p3 = __expf(st[3][t] - m_new);
      psum += (p0 + p1) + (p2 + p3);
      pf0[t]     = (_Float16)p0;
      pf0[t + 8] = (_Float16)p1;
      pf1[t]     = (_Float16)p2;
      pf1[t + 8] = (_Float16)p3;
    }
    l_run = l_run * scale + psum;

    // ---- O^T += V^T x P^T over 8 channel tiles (16 chans each) ----
#pragma unroll
    for (int t = 0; t < 8; ++t) {
#pragma unroll
      for (int e = 0; e < 8; ++e) acc[t][e] *= scale;
      const v16h vf0 = load_frag_16x32(Vb + (size_t)(c0 + 16 * t) * NTOK + j0,      NTOK, lane);
      acc[t] = __builtin_amdgcn_wmma_f32_16x16x32_f16(false, vf0, false, pf0, (short)0, acc[t], false, false);
      const v16h vf1 = load_frag_16x32(Vb + (size_t)(c0 + 16 * t) * NTOK + j0 + 32, NTOK, lane);
      acc[t] = __builtin_amdgcn_wmma_f32_16x16x32_f16(false, vf1, false, pf1, (short)0, acc[t], false, false);
    }
  }

  // ---- normalize + gamma-residual, store in [B, C, N] order ----
  // out^T tile layout == [B,C,N]: per VGPR the 32 lanes cover 2 channel rows
  // x 16 contiguous tokens (64B each) -> already coalesced, store direct.
  const float l_tot = l_run + __shfl_xor(l_run, 16, 32);
  const float rl = 1.0f / l_tot;
  const float g  = gamma[0];
  const int   i     = i0 + (lane & 15);
  const int   cbase = c0 + ((lane & 16) ? 8 : 0);
#pragma unroll
  for (int t = 0; t < 8; ++t) {
#pragma unroll
    for (int r = 0; r < 8; ++r) {
      const int c = cbase + 16 * t + r;
      const size_t idx = ((size_t)b * CH + c) * NTOK + i;
      out[idx] = g * (acc[t][r] * rl) + (1.0f - g) * x[idx];
    }
  }
}

extern "C" void kernel_launch(void* const* d_in, const int* in_sizes, int n_in,
                              void* d_out, int out_size, void* d_ws, size_t ws_size,
                              hipStream_t stream) {
  const float* x     = (const float*)d_in[0];
  const float* Wq    = (const float*)d_in[1];
  const float* bq    = (const float*)d_in[2];
  const float* Wk    = (const float*)d_in[3];
  const float* bk    = (const float*)d_in[4];
  const float* Wv    = (const float*)d_in[5];
  const float* bv    = (const float*)d_in[6];
  const float* gamma = (const float*)d_in[7];
  float* out = (float*)d_out;

  // Workspace layout (f16): Qh[B][N][64] (2MB) | Kh[B][N][64] (2MB) | Vt[B][512][N] (16MB)
  _Float16* Qh = (_Float16*)d_ws;
  _Float16* Kh = Qh + (size_t)BATCH * NTOK * DQK;
  _Float16* Vt = Kh + (size_t)BATCH * NTOK * DQK;

  dim3 qkgrid(NTOK / 64, BATCH);
  proj_qk_wmma<<<qkgrid, 128, 0, stream>>>(x, Wq, bq, Wk, bk, Qh, Kh);

  dim3 vgrid(NTOK / 64, CH / 128, BATCH);
  proj_v_wmma<<<vgrid, 128, 0, stream>>>(x, Wv, bv, Vt);

  // Attention: 256 query tiles x 4 batches, 4 waves/block (128 chans each)
  dim3 agrid(NTOK / 16, BATCH);
  attn_flash_kernel<<<agrid, 128, 0, stream>>>(Qh, Kh, Vt, x, gamma, out);
}
